// MixerModel_7782480741003
// MI455X (gfx1250) — compile-verified, matrix operand order checked by reading, not run on hardware
//
#include <hip/hip_runtime.h>
#include <hip/hip_bf16.h>

typedef __bf16 bf16_t;
typedef __attribute__((ext_vector_type(8)))  bf16_t v8bf;
typedef __attribute__((ext_vector_type(16))) bf16_t v16bf;
typedef __attribute__((ext_vector_type(8)))  float  v8f;

namespace {
constexpr int kB  = 2;
constexpr int kL  = 4096;
constexpr int kDM = 768;           // d_model
constexpr int kDI = 1536;          // d_inner
constexpr int kDS = 16;            // d_state
constexpr int kNL = 4;             // n_layer
constexpr int kM  = kB * kL;       // 8192 token rows
constexpr int kXZ = 2 * kDI;       // 3072
constexpr int kXPN = 96;           // x_proj N padded (dt_rank+2N = 80 -> 96)
constexpr int kDTK = 64;           // dt_proj K padded (48 -> 64)

constexpr size_t al256(size_t x) { return (x + 255) & ~size_t(255); }
// workspace layout (bytes)
constexpr size_t OFF_RES   = 0;
constexpr size_t OFF_H     = OFF_RES   + al256(4ull * kM * kDM);
constexpr size_t OFF_XZ    = OFF_H     + al256(4ull * kM * kDM);
constexpr size_t OFF_U     = OFF_XZ    + al256(4ull * kM * kXZ);
constexpr size_t OFF_XD    = OFF_U     + al256(4ull * kM * kDI);
constexpr size_t OFF_DELTA = OFF_XD    + al256(4ull * kM * kXPN);
constexpr size_t OFF_DT16  = OFF_DELTA + al256(4ull * kM * kDI);
constexpr size_t OFF_SH16  = OFF_DT16  + al256(2ull * kM * kDTK);   // hn16 / u16 / y16 (time-multiplexed)
constexpr size_t OFF_WIN   = OFF_SH16  + al256(2ull * kM * kDI);
constexpr size_t OFF_WXP   = OFF_WIN   + al256(2ull * kNL * kXZ * kDM);
constexpr size_t OFF_WDT   = OFF_WXP   + al256(2ull * kNL * kXPN * kDI);
constexpr size_t OFF_WOUT  = OFF_WDT   + al256(2ull * kNL * kDI * kDTK);
} // namespace

__device__ inline float softplusf(float x) {
  return x > 20.f ? x : log1pf(__expf(x));
}

// ---------------------------------------------------------------------------
// weight / activation conversion kernels (f32 -> bf16, with zero padding)
// ---------------------------------------------------------------------------
__global__ void f2bf_kernel(const float* __restrict__ in, bf16_t* __restrict__ out, int n) {
  int i = blockIdx.x * 256 + threadIdx.x;
  if (i < n) out[i] = (bf16_t)in[i];
}

// x_proj_W: (nl, 80, kDI) f32 -> (nl, 96, kDI) bf16, rows 80..95 zero
__global__ void cvt_xpw_kernel(const float* __restrict__ in, bf16_t* __restrict__ out) {
  int i = blockIdx.x * 256 + threadIdx.x;          // over kNL*kXPN*kDI
  int c  = i % kDI;
  int rg = i / kDI;
  int l  = rg / kXPN;
  int r  = rg % kXPN;
  out[i] = (r < 80) ? (bf16_t)in[((long)l * 80 + r) * kDI + c] : (bf16_t)0.f;
}

// dt_proj_W: (nl, kDI, 48) f32 -> (nl, kDI, 64) bf16, cols 48..63 zero
__global__ void cvt_dtw_kernel(const float* __restrict__ in, bf16_t* __restrict__ out) {
  int i = blockIdx.x * 256 + threadIdx.x;          // over kNL*kDI*kDTK
  int c   = i % kDTK;
  int row = i / kDTK;
  out[i] = (c < 48) ? (bf16_t)in[(long)row * 48 + c] : (bf16_t)0.f;
}

// x_dbl (kM, 96) f32 -> dt16 (kM, 64) bf16, cols 48..63 zero
__global__ void pad_dt16_kernel(const float* __restrict__ xdbl, bf16_t* __restrict__ dt16) {
  int i = blockIdx.x * 256 + threadIdx.x;          // over kM*kDTK
  int c   = i % kDTK;
  int row = i / kDTK;
  dt16[i] = (c < 48) ? (bf16_t)xdbl[(long)row * kXPN + c] : (bf16_t)0.f;
}

// ---------------------------------------------------------------------------
// embed: h[b,l,d] = sum_k x[b,k,l] * W[d,k] + bias[d]    grid=(3, kM), block=256
// ---------------------------------------------------------------------------
__global__ void embed_kernel(const float* __restrict__ x, const float* __restrict__ W,
                             const float* __restrict__ bias, float* __restrict__ h) {
  int d   = blockIdx.x * 256 + threadIdx.x;        // < 768
  int row = blockIdx.y;                            // (b,l)
  int b = row >> 12;
  int l = row & (kL - 1);
  float acc = bias[d];
  const float* xr = x + (long)b * 64 * kL + l;
  const float* wr = W + (long)d * 64;
#pragma unroll 8
  for (int k = 0; k < 64; ++k) acc += xr[(long)k * kL] * wr[k];
  h[(long)row * kDM + d] = acc;
}

// ---------------------------------------------------------------------------
// residual add + layernorm; writes updated residual and bf16 (or f32) output
// grid = kM blocks, 256 threads, 3 elements/thread (768)
// ---------------------------------------------------------------------------
__global__ void add_ln_kernel(const float* __restrict__ h, float* __restrict__ res,
                              const float* __restrict__ w, const float* __restrict__ bias,
                              bf16_t* __restrict__ out16, float* __restrict__ outf, int first) {
  __shared__ float sm[256];
  int row = blockIdx.x, t = threadIdx.x;
  const float* hr = h + (long)row * kDM;
  float* rr = res + (long)row * kDM;
  float v[3];
  float s = 0.f;
#pragma unroll
  for (int i = 0; i < 3; ++i) {
    int c = t + i * 256;
    float x = hr[c];
    if (!first) x += rr[c];
    rr[c] = x;
    v[i] = x;
    s += x;
  }
  sm[t] = s; __syncthreads();
  for (int k = 128; k > 0; k >>= 1) { if (t < k) sm[t] += sm[t + k]; __syncthreads(); }
  float mean = sm[0] / kDM; __syncthreads();
  float s2 = 0.f;
#pragma unroll
  for (int i = 0; i < 3; ++i) { float d = v[i] - mean; s2 += d * d; }
  sm[t] = s2; __syncthreads();
  for (int k = 128; k > 0; k >>= 1) { if (t < k) sm[t] += sm[t + k]; __syncthreads(); }
  float inv = rsqrtf(sm[0] / kDM + 1e-5f);
#pragma unroll
  for (int i = 0; i < 3; ++i) {
    int c = t + i * 256;
    float o = (v[i] - mean) * inv * w[c] + bias[c];
    if (out16) out16[(long)row * kDM + c] = (bf16_t)o;
    else       outf[(long)row * kDM + c] = o;
  }
}

// ---------------------------------------------------------------------------
// bf16 WMMA GEMM: C[M,N] = A[M,K] * Bw[N,K]^T  (both row-major over K)
// one wave -> 16x32 C tile (2 WMMA accumulators, A-fragment reuse)
// mode 1: fused   C = softplus(C + bias[n])   (dt_proj epilogue)
// ---------------------------------------------------------------------------
__device__ inline v16bf ld_frag(const bf16_t* p) {
  v8bf lo = *(const v8bf*)p;
  v8bf hi = *(const v8bf*)(p + 16);
  return __builtin_shufflevector(lo, hi, 0, 1, 2, 3, 4, 5, 6, 7,
                                 8, 9, 10, 11, 12, 13, 14, 15);
}

__global__ __launch_bounds__(128)
void gemm_wmma_bf16(const bf16_t* __restrict__ A, const bf16_t* __restrict__ Bw,
                    float* __restrict__ C, int Mt, int Nt2, int N, int K,
                    int mode, const float* __restrict__ bias) {
  int tile = blockIdx.x * 4 + threadIdx.y;         // 4 waves / block
  int tm = tile / Nt2;
  int tn = tile - tm * Nt2;
  if (tm >= Mt) return;                            // wave-uniform exit
  int lane = threadIdx.x;
  int half = lane >> 4;                            // K-half selector
  int r    = lane & 15;                            // M row (A) / N col (B,C)
  int m0 = tm * 16, n0 = tn * 32;

  const bf16_t* pa  = A  + (long)(m0 + r) * K + half * 8;
  const bf16_t* pb0 = Bw + (long)(n0 + r) * K + half * 8;
  const bf16_t* pb1 = pb0 + (long)16 * K;

  v8f acc0 = {0.f, 0.f, 0.f, 0.f, 0.f, 0.f, 0.f, 0.f};
  v8f acc1 = {0.f, 0.f, 0.f, 0.f, 0.f, 0.f, 0.f, 0.f};

  for (int k0 = 0; k0 < K; k0 += 32) {
    __builtin_prefetch(pa  + k0 + 256, 0, 1);      // global_prefetch_b8
    __builtin_prefetch(pb0 + k0 + 256, 0, 1);
    __builtin_prefetch(pb1 + k0 + 256, 0, 1);
    v16bf a  = ld_frag(pa  + k0);
    v16bf b0 = ld_frag(pb0 + k0);
    v16bf b1 = ld_frag(pb1 + k0);
    acc0 = __builtin_amdgcn_wmma_f32_16x16x32_bf16(false, a, false, b0,
                                                   (short)0, acc0, false, false);
    acc1 = __builtin_amdgcn_wmma_f32_16x16x32_bf16(false, a, false, b1,
                                                   (short)0, acc1, false, false);
  }

#pragma unroll
  for (int i = 0; i < 8; ++i) {
    int rowm = m0 + half * 8 + i;                  // C layout: vgpr i -> M = half*8 + i
    long idx = (long)rowm * N + n0 + r;
    float v0 = acc0[i], v1 = acc1[i];
    if (mode == 1) {
      v0 = softplusf(v0 + bias[n0 + r]);
      v1 = softplusf(v1 + bias[n0 + 16 + r]);
    }
    C[idx]      = v0;
    C[idx + 16] = v1;
  }
}

// ---------------------------------------------------------------------------
// depthwise causal conv1d (k=4) + SiLU; writes f32 u (for scan) and bf16 u (for x_proj)
// grid = (kDI/256, kM), block = 256
// ---------------------------------------------------------------------------
__global__ void conv_silu_kernel(const float* __restrict__ xz, const float* __restrict__ cw,
                                 const float* __restrict__ cb, float* __restrict__ u,
                                 bf16_t* __restrict__ u16) {
  int d   = blockIdx.x * 256 + threadIdx.x;        // channel in x-half
  int row = blockIdx.y;                            // (b,l)
  int l = row & (kL - 1);
  float acc = cb[d];
#pragma unroll
  for (int t = 0; t < 4; ++t) {
    int ls = l - 3 + t;
    if (ls >= 0) acc += xz[(long)(row - 3 + t) * kXZ + d] * cw[d * 4 + t];
  }
  float s = acc / (1.f + __expf(-acc));            // silu
  u[(long)row * kDI + d]   = s;
  u16[(long)row * kDI + d] = (bf16_t)s;
}

// ---------------------------------------------------------------------------
// selective scan: 16 lanes per (b,d) pair, one lane per state n; butterfly
// reduction over the 16 states; fused  y = (sum + D*u) * silu(z)  -> bf16
// grid = (kB*kDI*16)/256 blocks, 256 threads
// ---------------------------------------------------------------------------
__global__ void ssm_scan_kernel(const float* __restrict__ delta, const float* __restrict__ u,
                                const float* __restrict__ xdbl, const float* __restrict__ xz,
                                const float* __restrict__ Alog, const float* __restrict__ Dv,
                                bf16_t* __restrict__ y16) {
  int gt = blockIdx.x * 256 + threadIdx.x;
  int g = gt >> 4;                                 // (b,d) group
  int n = gt & 15;                                 // state index
  int b = g / kDI;
  int d = g - b * kDI;
  float An = -__expf(Alog[d * kDS + n]);
  float Dd = Dv[d];
  float s = 0.f;
  long base = (long)b * kL;
  for (int l = 0; l < kL; ++l) {
    long row = base + l;
    float dl = delta[row * kDI + d];
    float uu = u[row * kDI + d];
    float Bn = xdbl[row * kXPN + 48 + n];
    float Cn = xdbl[row * kXPN + 64 + n];
    float dA = __expf(dl * An);
    s = dA * s + (dl * uu) * Bn;
    float p = s * Cn;
    p += __shfl_xor(p, 1);
    p += __shfl_xor(p, 2);
    p += __shfl_xor(p, 4);
    p += __shfl_xor(p, 8);
    if (n == 0) {
      float z = xz[row * kXZ + kDI + d];
      float y = p + Dd * uu;
      y *= z / (1.f + __expf(-z));                 // * silu(z)
      y16[row * kDI + d] = (bf16_t)y;
    }
  }
}

// ---------------------------------------------------------------------------
extern "C" void kernel_launch(void* const* d_in, const int* in_sizes, int n_in,
                              void* d_out, int out_size, void* d_ws, size_t ws_size,
                              hipStream_t stream) {
  (void)in_sizes; (void)n_in; (void)out_size; (void)ws_size;
  const float* d_x     = (const float*)d_in[0];
  const float* d_embW  = (const float*)d_in[1];
  const float* d_embB  = (const float*)d_in[2];
  const float* d_normw = (const float*)d_in[3];
  const float* d_normb = (const float*)d_in[4];
  const float* d_inW   = (const float*)d_in[5];
  const float* d_convw = (const float*)d_in[6];
  const float* d_convb = (const float*)d_in[7];
  const float* d_xpW   = (const float*)d_in[8];
  const float* d_dtW   = (const float*)d_in[9];
  const float* d_dtb   = (const float*)d_in[10];
  const float* d_Alog  = (const float*)d_in[11];
  const float* d_Dv    = (const float*)d_in[12];
  const float* d_outW  = (const float*)d_in[13];
  const float* d_nfw   = (const float*)d_in[14];
  const float* d_nfb   = (const float*)d_in[15];

  char* ws = (char*)d_ws;
  float*  res    = (float*)(ws + OFF_RES);
  float*  h      = (float*)(ws + OFF_H);
  float*  xz     = (float*)(ws + OFF_XZ);
  float*  u      = (float*)(ws + OFF_U);
  float*  xdbl   = (float*)(ws + OFF_XD);
  float*  delta  = (float*)(ws + OFF_DELTA);
  bf16_t* dt16   = (bf16_t*)(ws + OFF_DT16);
  bf16_t* sh16   = (bf16_t*)(ws + OFF_SH16);     // hn16 / u16 / y16
  bf16_t* win16  = (bf16_t*)(ws + OFF_WIN);
  bf16_t* wxp16  = (bf16_t*)(ws + OFF_WXP);
  bf16_t* wdt16  = (bf16_t*)(ws + OFF_WDT);
  bf16_t* wout16 = (bf16_t*)(ws + OFF_WOUT);

  auto launch_gemm = [&](const bf16_t* A, const bf16_t* Bw, float* C,
                         int M, int N, int K, int mode, const float* bias) {
    int Mt = M / 16, Nt2 = N / 32;
    int total = Mt * Nt2;
    dim3 grid((total + 3) / 4), block(32, 4);
    gemm_wmma_bf16<<<grid, block, 0, stream>>>(A, Bw, C, Mt, Nt2, N, K, mode, bias);
  };

  // convert all weights to bf16 (padded where needed)
  f2bf_kernel<<<(kNL * kXZ * kDM) / 256, 256, 0, stream>>>(d_inW, win16, kNL * kXZ * kDM);
  cvt_xpw_kernel<<<(kNL * kXPN * kDI) / 256, 256, 0, stream>>>(d_xpW, wxp16);
  cvt_dtw_kernel<<<(kNL * kDI * kDTK) / 256, 256, 0, stream>>>(d_dtW, wdt16);
  f2bf_kernel<<<(kNL * kDM * kDI) / 256, 256, 0, stream>>>(d_outW, wout16, kNL * kDM * kDI);

  // embedding
  embed_kernel<<<dim3(kDM / 256, kM), 256, 0, stream>>>(d_x, d_embW, d_embB, h);

  for (int i = 0; i < kNL; ++i) {
    // residual add + layernorm -> bf16 hn (in sh16)
    add_ln_kernel<<<kM, 256, 0, stream>>>(h, res, d_normw + i * kDM, d_normb + i * kDM,
                                          sh16, nullptr, i == 0 ? 1 : 0);
    // in_proj: (kM,768) x (3072,768)^T -> xz (kM,3072)
    launch_gemm(sh16, win16 + (size_t)i * kXZ * kDM, xz, kM, kXZ, kDM, 0, nullptr);
    // depthwise conv + silu -> u (f32) + u16 (bf16, reuses sh16)
    conv_silu_kernel<<<dim3(kDI / 256, kM), 256, 0, stream>>>(
        xz, d_convw + (size_t)i * kDI * 4, d_convb + (size_t)i * kDI, u, sh16);
    // x_proj: (kM,1536) x (96,1536)^T -> x_dbl (kM,96)
    launch_gemm(sh16, wxp16 + (size_t)i * kXPN * kDI, xdbl, kM, kXPN, kDI, 0, nullptr);
    // dt slice -> bf16 padded (kM,64)
    pad_dt16_kernel<<<(kM * kDTK) / 256, 256, 0, stream>>>(xdbl, dt16);
    // dt_proj + fused bias + softplus -> delta (kM,1536)
    launch_gemm(dt16, wdt16 + (size_t)i * kDI * kDTK, delta, kM, kDI, kDTK, 1,
                d_dtb + (size_t)i * kDI);
    // selective scan -> y16 (reuses sh16)
    ssm_scan_kernel<<<(kB * kDI * 16) / 256, 256, 0, stream>>>(
        delta, u, xdbl, xz, d_Alog + (size_t)i * kDI * kDS, d_Dv + (size_t)i * kDI, sh16);
    // out_proj: (kM,1536) x (768,1536)^T -> h (kM,768)
    launch_gemm(sh16, wout16 + (size_t)i * kDM * kDI, h, kM, kDM, kDI, 0, nullptr);
  }

  // final residual add + layernorm -> d_out (f32)
  add_ln_kernel<<<kM, 256, 0, stream>>>(h, res, d_nfw, d_nfb, nullptr, (float*)d_out, 0);
}
